// STORN_73031623901799
// MI455X (gfx1250) — matrix operand order repaired and checked by reading, not compiled, not run on hardware
//
#include <hip/hip_runtime.h>

// ---------------------------------------------------------------------------
// STORN loss on gfx1250 (MI455X), wave32 + V_WMMA_F32_16X16X32_F16.
//
// Roofline: the scan is sequentially dependent; per-step bytes are dominated
// by the f16 weights (~17MB) which fit easily in the 192MB L2, so the scan is
// L2-resident (one-time ~100MB HBM traffic vs 23.3TB/s). f16 WMMA with f32
// accumulation halves weight traffic vs fp32 and runs on the matrix pipe;
// all reductions/elementwise math stay fp32.
//
// CDNA5-specific paths exercised:
//   * v_wmma_f32_16x16x32_f16 for every GEMM (A frag: split 8-half chunks per
//     lane-half; B frag: 16 contiguous halves, W stored [N,K] row-major).
//   * global_load_async_to_lds_b128 + s_wait_asynccnt to stage W0 in k_phi.
//   * ds_load_b128 A-fragments from LDS, ds_add_f32 loss reduction.
//   * global_prefetch_b8 (__builtin_prefetch) to pipeline next-stage weights.
// ---------------------------------------------------------------------------

constexpr int kB = 128;    // batch
constexpr int kT = 256;    // seq
constexpr int kH = 512;    // hidden
constexpr int kZ = 64;     // latent
constexpr int kY = 16;     // obs
constexpr int kG = 1536;   // 3*H (GRU gates)

typedef __attribute__((ext_vector_type(16))) _Float16 v16h;
typedef __attribute__((ext_vector_type(8)))  _Float16 v8h;
typedef __attribute__((ext_vector_type(8)))  float    v8f;
typedef __attribute__((ext_vector_type(4)))  int      v4i;

// ---- WMMA fragment loaders (wave32; k order contiguous in A and B, m/n lane map per ISA) ----------
// A (16x32 f16): lanes 0-15 hold K = k0+0..15 for row m0+ln;
//                lanes 16-31 hold K = k0+16..31 (contiguous, matches B).
__device__ __forceinline__ v16h ld_a(const _Float16* A, int lda, int m0,
                                     int k0, int lane) {
  const int hi = (lane >> 4) & 1, ln = lane & 15;
  const _Float16* p = A + (size_t)(m0 + ln) * lda + k0 + hi * 16;
  union { v16h v; v8h h[2]; } u;
  u.h[0] = *(const v8h*)p;
  u.h[1] = *(const v8h*)(p + 8);
  return u.v;
}
// B (32x16 f16), W stored row-major [N,K] (so B = W^T): lane -> column n0+ln,
// lanes 0-15 hold K=k0..k0+15, lanes 16-31 hold K=k0+16..k0+31 (contiguous).
__device__ __forceinline__ v16h ld_b(const _Float16* W, int ldw, int n0,
                                     int k0, int lane) {
  const int hi = (lane >> 4) & 1, ln = lane & 15;
  return *(const v16h*)(W + (size_t)(n0 + ln) * ldw + k0 + hi * 16);
}

// ---- small utility kernels ------------------------------------------------
__global__ void k_zero(unsigned int* p, long n) {
  long i = (long)blockIdx.x * blockDim.x + threadIdx.x;
  if (i < n) p[i] = 0u;
}

// fp32 [N,K] row-major -> f16 [N,Kp] row-major with zero pad K..Kp
__global__ void k_cvt_w(const float* __restrict__ src, _Float16* __restrict__ dst,
                        int N, int K, int Kp) {
  int i = blockIdx.x * blockDim.x + threadIdx.x;
  if (i >= N * Kp) return;
  int n = i / Kp, k = i - n * Kp;
  dst[i] = (k < K) ? (_Float16)src[(size_t)n * K + k] : (_Float16)0.0f;
}

// src [B, C, T] fp32 -> dst [(t*B+b)*32 + c] f16, zero pad c in [C,32)
__global__ void k_stage(const float* __restrict__ src, _Float16* __restrict__ dst,
                        int C) {
  int i = blockIdx.x * blockDim.x + threadIdx.x;
  if (i >= kT * kB * 32) return;
  int c = i & 31;
  int b = (i >> 5) & (kB - 1);
  int t = i >> 12;  // 32*128 = 4096
  dst[i] = (c < C) ? (_Float16)src[((size_t)b * C + c) * kT + t] : (_Float16)0.0f;
}

// ---- fused 2-layer MLP (Linear->ReLU->Linear) over all T*B rows ------------
// One block = 16 rows. W0 (32KB) is staged once per block into LDS with
// async-to-LDS loads (ASYNCcnt); layer-0 output staged in LDS and re-read
// with ds_load_b128 as WMMA fragments for layer 1.
__global__ __launch_bounds__(256) void k_phi(const _Float16* __restrict__ X,
                                             const _Float16* __restrict__ W0,
                                             const float* __restrict__ b0,
                                             const _Float16* __restrict__ W1,
                                             const float* __restrict__ b1,
                                             _Float16* __restrict__ out) {
  __shared__ _Float16 w0s[kH * 32];  // 32 KB: full W0 [512,32] f16
  __shared__ _Float16 mid[16][520];  // pad stride: 1040B rows, 16B aligned
  const int tid = threadIdx.x, wave = tid >> 5, lane = tid & 31;
  const int hi = lane >> 4, ln = lane & 15;
  const size_t rowbase = (size_t)blockIdx.x * 16;

  // stage W0 -> LDS (2048 x 16B chunks over 256 threads)
#if __has_builtin(__builtin_amdgcn_global_load_async_to_lds_b128)
  for (int i = tid; i < kH * 32 / 8; i += 256) {
    __attribute__((address_space(1))) v4i* gsrc =
        (__attribute__((address_space(1))) v4i*)(unsigned long long)
            (const void*)(W0 + (size_t)i * 8);
    __attribute__((address_space(3))) v4i* ldst =
        (__attribute__((address_space(3))) v4i*)(unsigned)
            (unsigned long long)(void*)(&w0s[i * 8]);
    __builtin_amdgcn_global_load_async_to_lds_b128(gsrc, ldst, 0, 0);
  }
#if __has_builtin(__builtin_amdgcn_s_wait_asynccnt)
  __builtin_amdgcn_s_wait_asynccnt(0);
#else
  asm volatile("s_wait_asynccnt 0x0" ::: "memory");
#endif
#else
  for (int i = tid; i < kH * 32 / 8; i += 256)
    *(v8h*)(&w0s[i * 8]) = *(const v8h*)(W0 + (size_t)i * 8);
#endif
  __syncthreads();

  // layer 0: K=32 (padded), N=512 -> 32 tiles over 8 waves; B from LDS
  for (int n0 = wave * 16; n0 < kH; n0 += 128) {
    union { v16h v; v8h h[2]; } ua;
    const _Float16* pa = X + (rowbase + ln) * 32 + hi * 16;
    ua.h[0] = *(const v8h*)pa;
    ua.h[1] = *(const v8h*)(pa + 8);
    union { v16h v; v8h h[2]; } ub;
    const _Float16* pb = &w0s[(n0 + ln) * 32 + hi * 16];
    ub.h[0] = *(const v8h*)pb;
    ub.h[1] = *(const v8h*)(pb + 8);
    v8f acc = {};
    acc = __builtin_amdgcn_wmma_f32_16x16x32_f16(false, ua.v, false, ub.v,
                                                 (short)0, acc, false, false);
    const float bv = b0[n0 + ln];
#pragma unroll
    for (int r = 0; r < 8; ++r)
      mid[r + hi * 8][n0 + ln] = (_Float16)fmaxf(acc[r] + bv, 0.0f);
  }
  __syncthreads();

  // layer 1: K=512, N=512 ; A fragments from LDS
  for (int n0 = wave * 16; n0 < kH; n0 += 128) {
    v8f acc = {};
    for (int k = 0; k < kH; k += 32) {
      union { v16h v; v8h h[2]; } ua;
      ua.h[0] = *(const v8h*)(&mid[ln][k + hi * 16]);
      ua.h[1] = *(const v8h*)(&mid[ln][k + hi * 16 + 8]);
      v16h b = *(const v16h*)(W1 + (size_t)(n0 + ln) * kH + k + hi * 16);
      acc = __builtin_amdgcn_wmma_f32_16x16x32_f16(false, ua.v, false, b,
                                                   (short)0, acc, false, false);
    }
    const float bv = b1[n0 + ln];
#pragma unroll
    for (int r = 0; r < 8; ++r)
      out[(rowbase + r + hi * 8) * kH + n0 + ln] = (_Float16)(acc[r] + bv);
  }
}

// ---- per-timestep kernel ---------------------------------------------------
struct StepArgs {
  const _Float16 *PHIY, *PHIU;
  const float *eps, *y;
  const _Float16 *WI0I, *WI0H, *WI1I, *WI1H;     // rnn_inf
  const _Float16 *WG0I, *WG0H, *WG1I, *WG1H;     // rnn_gen
  const _Float16 *WE0, *WE1, *WEM, *WEL;         // enc + heads
  const _Float16 *WZ0, *WZ1, *WD0, *WD1, *WDM, *WDL;
  const float *bE0, *bE1, *bEM, *bEL, *bZ0, *bZ1, *bD0, *bD1, *bDM, *bDL;
  float *H0, *H1, *D0, *D1, *GX, *GH, *MUQ, *LQ, *MUP, *LP;
  _Float16 *H0h, *H1h, *D0h, *D1h, *E0h, *E1h, *Zh, *P0h, *XG, *DEC0h, *DEC1h;
  float* out;
  int t;
};

__global__ __launch_bounds__(1024) void k_step(StepArgs A) {
  const int tid = threadIdx.x;
  const int wave = tid >> 5;
  const int lane = tid & 31;
  const int hi = lane >> 4, ln = lane & 15;
  float lloss = 0.0f;
  const _Float16* phiy_t = A.PHIY + (size_t)A.t * kB * kH;
  const _Float16* phiu_t = A.PHIU + (size_t)A.t * kB * kH;
  float* GX = A.GX;
  float* GH = A.GH;

  // prefetch of a weight matrix into WGP$/L2 (global_prefetch_b8), capped at
  // <= 8 cachelines per thread so it is ~free next to a GEMM stage.
  auto pf = [&](const void* p, size_t bytes) {
    const char* c = (const char*)p;
    size_t cap = (size_t)1024 * 128 * 8;
    if (bytes > cap) bytes = cap;
    for (size_t o = (size_t)tid * 128; o < bytes; o += (size_t)1024 * 128)
      __builtin_prefetch(c + o, 0, 0);
  };

  // O[M=128,N] = Act @ W^T (+bias, relu); M tiles = 8, distributed on 32 waves
  auto run_gemm = [&](const _Float16* Am, int K, const _Float16* W, float* Of,
                      _Float16* Oh, int ldo, const float* bias, bool relu,
                      int N) {
    const int ntiles = (N >> 4) << 3;
    for (int tix = wave; tix < ntiles; tix += 32) {
      const int m0 = (tix & 7) << 4;
      const int n0 = (tix >> 3) << 4;
      v8f acc = {};
      for (int k = 0; k < K; k += 32)
        acc = __builtin_amdgcn_wmma_f32_16x16x32_f16(
            false, ld_a(Am, K, m0, k, lane), false, ld_b(W, K, n0, k, lane),
            (short)0, acc, false, false);
      const float bv = bias ? bias[n0 + ln] : 0.0f;
#pragma unroll
      for (int r = 0; r < 8; ++r) {
        float v = acc[r] + bv;
        if (relu) v = fmaxf(v, 0.0f);
        const size_t o = (size_t)(m0 + r + hi * 8) * ldo + n0 + ln;
        if (Of) Of[o] = v;
        if (Oh) Oh[o] = (_Float16)v;
      }
    }
  };

  // GRU gate fusion from GX/GH [128,1536]; in-place state + f16 mirror
  auto gru_gate = [&](float* Hs, _Float16* Hm) {
    for (int i = tid; i < kB * kH; i += 1024) {
      const int b = i >> 9, j = i & (kH - 1);
      const float* gx = GX + (size_t)b * kG;
      const float* gh = GH + (size_t)b * kG;
      const float r = 1.0f / (1.0f + expf(-(gx[j] + gh[j])));
      const float zg = 1.0f / (1.0f + expf(-(gx[kH + j] + gh[kH + j])));
      const float n = tanhf(gx[2 * kH + j] + r * gh[2 * kH + j]);
      const float hv = (1.0f - zg) * n + zg * Hs[i];
      Hs[i] = hv;
      Hm[i] = (_Float16)hv;
    }
  };

  // -- inference GRU, layer 0 (prefetch layer-1 weights while it runs)
  pf(A.WI1I, (size_t)kG * kH * 2);
  run_gemm(phiy_t, kH, A.WI0I, GX, nullptr, kG, nullptr, false, kG);
  run_gemm(A.D0h, kH, A.WI0H, GH, nullptr, kG, nullptr, false, kG);
  __syncthreads();
  gru_gate(A.D0, A.D0h);
  __syncthreads();
  // -- inference GRU, layer 1
  pf(A.WE0, (size_t)kH * kH * 2);
  run_gemm(A.D0h, kH, A.WI1I, GX, nullptr, kG, nullptr, false, kG);
  run_gemm(A.D1h, kH, A.WI1H, GH, nullptr, kG, nullptr, false, kG);
  __syncthreads();
  gru_gate(A.D1, A.D1h);
  __syncthreads();
  // -- encoder MLP (relu, relu); prefetch the big gen-GRU input weights
  pf(A.WG0I, (size_t)kG * 2 * kH * 2);
  run_gemm(A.D1h, kH, A.WE0, nullptr, A.E0h, kH, A.bE0, true, kH);
  __syncthreads();
  run_gemm(A.E0h, kH, A.WE1, nullptr, A.E1h, kH, A.bE1, true, kH);
  __syncthreads();
  // -- heads: mu_q, relu(logvar_q); also copy phi_u into XG[:, :512]
  run_gemm(A.E1h, kH, A.WEM, A.MUQ, nullptr, kZ, A.bEM, false, kZ);
  run_gemm(A.E1h, kH, A.WEL, A.LQ, nullptr, kZ, A.bEL, true, kZ);
  for (int i = tid; i < kB * kH; i += 1024)
    A.XG[(size_t)(i >> 9) * (2 * kH) + (i & (kH - 1))] = phiu_t[i];
  __syncthreads();
  // -- reparameterized sample + KLD
  {
    const float* eps_t = A.eps + (size_t)A.t * kB * kZ;
    for (int i = tid; i < kB * kZ; i += 1024) {
      const float mu = A.MUQ[i], lv = A.LQ[i];
      A.Zh[i] = (_Float16)(mu + expf(0.5f * lv) * eps_t[i]);
      lloss += 0.5f * (-lv - 1.0f + expf(lv) + mu * mu);
    }
  }
  __syncthreads();
  // -- phi_z MLP (relu, none); layer-1 writes straight into XG[:, 512:]
  run_gemm(A.Zh, kZ, A.WZ0, nullptr, A.P0h, kH, A.bZ0, true, kH);
  __syncthreads();
  run_gemm(A.P0h, kH, A.WZ1, nullptr, A.XG + kH, 2 * kH, A.bZ1, false, kH);
  // -- decoder from h[-1] (prev h1), independent of phi_z
  run_gemm(A.H1h, kH, A.WD0, nullptr, A.DEC0h, kH, A.bD0, true, kH);
  __syncthreads();
  run_gemm(A.DEC0h, kH, A.WD1, nullptr, A.DEC1h, kH, A.bD1, false, kH);
  __syncthreads();
  run_gemm(A.DEC1h, kH, A.WDM, A.MUP, nullptr, kY, A.bDM, false, kY);
  run_gemm(A.DEC1h, kH, A.WDL, A.LP, nullptr, kY, A.bDL, true, kY);
  __syncthreads();
  // -- Gaussian log-likelihood of y_t (loss accumulates -logprob)
  for (int i = tid; i < kB * kY; i += 1024) {
    const int b = i >> 4, c = i & 15;
    const float yv = A.y[((size_t)b * kY + c) * kT + A.t];
    const float df = yv - A.MUP[i];
    const float lp = A.LP[i];
    lloss += 0.5f * df * df * expf(-lp) + 0.5f * lp + 0.91893853320467274f;
  }
  // -- generative GRU, layer 0 (input K = 1024 = [phi_u ; phi_z])
  pf(A.WG1I, (size_t)kG * kH * 2);
  run_gemm(A.XG, 2 * kH, A.WG0I, GX, nullptr, kG, nullptr, false, kG);
  run_gemm(A.H0h, kH, A.WG0H, GH, nullptr, kG, nullptr, false, kG);
  __syncthreads();
  gru_gate(A.H0, A.H0h);
  __syncthreads();
  // -- generative GRU, layer 1
  run_gemm(A.H0h, kH, A.WG1I, GX, nullptr, kG, nullptr, false, kG);
  run_gemm(A.H1h, kH, A.WG1H, GH, nullptr, kG, nullptr, false, kG);
  __syncthreads();
  gru_gate(A.H1, A.H1h);
  // -- reduce loss: LDS ds_add_f32 then one global atomic
  __shared__ float lsum;
  if (tid == 0) lsum = 0.0f;
  __syncthreads();
  atomicAdd(&lsum, lloss);
  __syncthreads();
  if (tid == 0) atomicAdd(A.out, lsum);
}

// ---------------------------------------------------------------------------
extern "C" void kernel_launch(void* const* d_in, const int* in_sizes, int n_in,
                              void* d_out, int out_size, void* d_ws,
                              size_t ws_size, hipStream_t stream) {
  (void)in_sizes; (void)n_in; (void)ws_size;
  const float* u = (const float*)d_in[0];
  const float* y = (const float*)d_in[1];
  const float* eps = (const float*)d_in[2];

  char* cur = (char*)d_ws;
  auto bump = [&](size_t bytes) -> char* {
    char* p = cur;
    cur += (bytes + 255) & ~(size_t)255;
    return p;
  };
  auto cvt = [&](int idx, _Float16* dst, int N, int K, int Kp) {
    const int n = N * Kp;
    k_cvt_w<<<(n + 255) / 256, 256, 0, stream>>>((const float*)d_in[idx], dst,
                                                 N, K, Kp);
  };

  // ---- f16 weight pool (params flattened in reference dict order, W then b)
  _Float16* WY0 = (_Float16*)bump((size_t)512 * 32 * 2);     // phi_y[0], K 16->32
  _Float16* WY1 = (_Float16*)bump((size_t)512 * 512 * 2);
  _Float16* WU0 = (_Float16*)bump((size_t)512 * 32 * 2);
  _Float16* WU1 = (_Float16*)bump((size_t)512 * 512 * 2);
  _Float16* WZ0 = (_Float16*)bump((size_t)512 * 64 * 2);
  _Float16* WZ1 = (_Float16*)bump((size_t)512 * 512 * 2);
  _Float16* WE0 = (_Float16*)bump((size_t)512 * 512 * 2);
  _Float16* WE1 = (_Float16*)bump((size_t)512 * 512 * 2);
  _Float16* WEM = (_Float16*)bump((size_t)64 * 512 * 2);
  _Float16* WEL = (_Float16*)bump((size_t)64 * 512 * 2);
  _Float16* WD0 = (_Float16*)bump((size_t)512 * 512 * 2);
  _Float16* WD1 = (_Float16*)bump((size_t)512 * 512 * 2);
  _Float16* WDM = (_Float16*)bump((size_t)16 * 512 * 2);
  _Float16* WDL = (_Float16*)bump((size_t)16 * 512 * 2);
  _Float16* WG0I = (_Float16*)bump((size_t)1536 * 1024 * 2);
  _Float16* WG0H = (_Float16*)bump((size_t)1536 * 512 * 2);
  _Float16* WG1I = (_Float16*)bump((size_t)1536 * 512 * 2);
  _Float16* WG1H = (_Float16*)bump((size_t)1536 * 512 * 2);
  _Float16* WI0I = (_Float16*)bump((size_t)1536 * 512 * 2);
  _Float16* WI0H = (_Float16*)bump((size_t)1536 * 512 * 2);
  _Float16* WI1I = (_Float16*)bump((size_t)1536 * 512 * 2);
  _Float16* WI1H = (_Float16*)bump((size_t)1536 * 512 * 2);

  cvt(3, WY0, 512, 16, 32);   cvt(5, WY1, 512, 512, 512);
  cvt(7, WU0, 512, 32, 32);   cvt(9, WU1, 512, 512, 512);
  cvt(11, WZ0, 512, 64, 64);  cvt(13, WZ1, 512, 512, 512);
  cvt(15, WE0, 512, 512, 512); cvt(17, WE1, 512, 512, 512);
  cvt(19, WEM, 64, 512, 512);  cvt(21, WEL, 64, 512, 512);
  cvt(23, WD0, 512, 512, 512); cvt(25, WD1, 512, 512, 512);
  cvt(27, WDM, 16, 512, 512);  cvt(29, WDL, 16, 512, 512);
  cvt(31, WG0I, 1536, 1024, 1024); cvt(32, WG0H, 1536, 512, 512);
  cvt(33, WG1I, 1536, 512, 512);   cvt(34, WG1H, 1536, 512, 512);
  cvt(35, WI0I, 1536, 512, 512);   cvt(36, WI0H, 1536, 512, 512);
  cvt(37, WI1I, 1536, 512, 512);   cvt(38, WI1H, 1536, 512, 512);

  const float *bY0 = (const float*)d_in[4], *bY1 = (const float*)d_in[6];
  const float *bU0 = (const float*)d_in[8], *bU1 = (const float*)d_in[10];
  const float *bZ0 = (const float*)d_in[12], *bZ1 = (const float*)d_in[14];
  const float *bE0 = (const float*)d_in[16], *bE1 = (const float*)d_in[18];
  const float *bEM = (const float*)d_in[20], *bEL = (const float*)d_in[22];
  const float *bD0 = (const float*)d_in[24], *bD1 = (const float*)d_in[26];
  const float *bDM = (const float*)d_in[28], *bDL = (const float*)d_in[30];

  // ---- staged inputs + batched phi tensors
  _Float16* Yst = (_Float16*)bump((size_t)kT * kB * 32 * 2);
  _Float16* Ust = (_Float16*)bump((size_t)kT * kB * 32 * 2);
  _Float16* PHIY = (_Float16*)bump((size_t)kT * kB * kH * 2);
  _Float16* PHIU = (_Float16*)bump((size_t)kT * kB * kH * 2);
  {
    const int n = kT * kB * 32;
    k_stage<<<(n + 255) / 256, 256, 0, stream>>>(y, Yst, 16);
    k_stage<<<(n + 255) / 256, 256, 0, stream>>>(u, Ust, 32);
  }
  k_phi<<<(kT * kB) / 16, 256, 0, stream>>>(Yst, WY0, bY0, WY1, bY1, PHIY);
  k_phi<<<(kT * kB) / 16, 256, 0, stream>>>(Ust, WU0, bU0, WU1, bU1, PHIU);

  // ---- scan state (zeroed each call) + per-step temporaries
  float* H0 = (float*)bump((size_t)kB * kH * 4);
  float* H1 = (float*)bump((size_t)kB * kH * 4);
  float* D0 = (float*)bump((size_t)kB * kH * 4);
  float* D1 = (float*)bump((size_t)kB * kH * 4);
  _Float16* H0h = (_Float16*)bump((size_t)kB * kH * 2);
  _Float16* H1h = (_Float16*)bump((size_t)kB * kH * 2);
  _Float16* D0h = (_Float16*)bump((size_t)kB * kH * 2);
  _Float16* D1h = (_Float16*)bump((size_t)kB * kH * 2);
  {
    const long zwords = (long)kB * kH * 24 / 4;  // 4x f32 + 4x f16 states
    k_zero<<<(int)((zwords + 255) / 256), 256, 0, stream>>>((unsigned*)H0,
                                                            zwords);
    k_zero<<<1, 256, 0, stream>>>((unsigned*)d_out, out_size);
  }
  float* GX = (float*)bump((size_t)kB * kG * 4);
  float* GH = (float*)bump((size_t)kB * kG * 4);
  float* MUQ = (float*)bump((size_t)kB * kZ * 4);
  float* LQ = (float*)bump((size_t)kB * kZ * 4);
  float* MUP = (float*)bump((size_t)kB * kY * 4);
  float* LP = (float*)bump((size_t)kB * kY * 4);
  _Float16* E0h = (_Float16*)bump((size_t)kB * kH * 2);
  _Float16* E1h = (_Float16*)bump((size_t)kB * kH * 2);
  _Float16* Zh = (_Float16*)bump((size_t)kB * kZ * 2);
  _Float16* P0h = (_Float16*)bump((size_t)kB * kH * 2);
  _Float16* XG = (_Float16*)bump((size_t)kB * 2 * kH * 2);
  _Float16* DEC0h = (_Float16*)bump((size_t)kB * kH * 2);
  _Float16* DEC1h = (_Float16*)bump((size_t)kB * kH * 2);

  StepArgs sa;
  sa.PHIY = PHIY; sa.PHIU = PHIU; sa.eps = eps; sa.y = y;
  sa.WI0I = WI0I; sa.WI0H = WI0H; sa.WI1I = WI1I; sa.WI1H = WI1H;
  sa.WG0I = WG0I; sa.WG0H = WG0H; sa.WG1I = WG1I; sa.WG1H = WG1H;
  sa.WE0 = WE0; sa.WE1 = WE1; sa.WEM = WEM; sa.WEL = WEL;
  sa.WZ0 = WZ0; sa.WZ1 = WZ1; sa.WD0 = WD0; sa.WD1 = WD1;
  sa.WDM = WDM; sa.WDL = WDL;
  sa.bE0 = bE0; sa.bE1 = bE1; sa.bEM = bEM; sa.bEL = bEL;
  sa.bZ0 = bZ0; sa.bZ1 = bZ1; sa.bD0 = bD0; sa.bD1 = bD1;
  sa.bDM = bDM; sa.bDL = bDL;
  sa.H0 = H0; sa.H1 = H1; sa.D0 = D0; sa.D1 = D1;
  sa.GX = GX; sa.GH = GH; sa.MUQ = MUQ; sa.LQ = LQ; sa.MUP = MUP; sa.LP = LP;
  sa.H0h = H0h; sa.H1h = H1h; sa.D0h = D0h; sa.D1h = D1h;
  sa.E0h = E0h; sa.E1h = E1h; sa.Zh = Zh; sa.P0h = P0h; sa.XG = XG;
  sa.DEC0h = DEC0h; sa.DEC1h = DEC1h;
  sa.out = (float*)d_out;

  for (int t = 0; t < kT; ++t) {
    sa.t = t;
    k_step<<<1, 1024, 0, stream>>>(sa);
  }
  (void)bY0; (void)bY1; (void)bU0; (void)bU1;
}
